// QwenAttention_12506944766692
// MI455X (gfx1250) — compile-verified, matrix operand order checked by reading, not run on hardware
//
#include <hip/hip_runtime.h>

// ---------------- problem constants ----------------
constexpr int Bq  = 2;
constexpr int Tq  = 2048;
constexpr int Dq  = 2048;
constexpr int Hq  = 16;
constexpr int Kq  = 8;
constexpr int Hd  = 128;
constexpr int BT  = Bq * Tq;          // 4096 rows
constexpr float RMS_EPS = 1e-6f;

typedef __attribute__((ext_vector_type(16))) __bf16 v16bf;
typedef __attribute__((ext_vector_type(8)))  float  v8f;

struct alignas(16) U4 { unsigned int v[4]; };

#if defined(__AMDGCN__) && __has_builtin(__builtin_amdgcn_global_load_async_to_lds_b128)
#define ASYNC_LDS 1
#else
#define ASYNC_LDS 0
#endif

#if ASYNC_LDS
typedef __attribute__((__vector_size__(4 * sizeof(int)))) int v4i_t;
typedef __attribute__((address_space(1))) v4i_t* gptr_v4i;
typedef __attribute__((address_space(3))) v4i_t* lptr_v4i;
#endif

__device__ inline void async_cp16(void* lds_dst, const void* gsrc) {
#if ASYNC_LDS
  // gfx1250: GLOBAL_LOAD_ASYNC_TO_LDS_B128 — no VGPR data, tracked by ASYNCcnt
  __builtin_amdgcn_global_load_async_to_lds_b128(
      (gptr_v4i)gsrc, (lptr_v4i)lds_dst, 0, 0);
#else
  *reinterpret_cast<U4*>(lds_dst) = *reinterpret_cast<const U4*>(gsrc);
#endif
}

__device__ inline void async_wait0() {
#if ASYNC_LDS
#if __has_builtin(__builtin_amdgcn_s_wait_asynccnt)
  __builtin_amdgcn_s_wait_asynccnt(0);
#else
  asm volatile("s_wait_asynccnt 0x0" ::: "memory");
#endif
#endif
}

__device__ inline v16bf load_frag16(const __bf16* p) {
  // CDNA5 16-bit A/B fragment: VGPR0-3 = K {base..base+7}, VGPR4-7 = K {base+16..base+23}
  v16bf f;
  U4* fp = reinterpret_cast<U4*>(&f);
  fp[0] = *reinterpret_cast<const U4*>(p);
  fp[1] = *reinterpret_cast<const U4*>(p + 16);
  return f;
}

__device__ inline v8f zero8() {
  v8f z = {0.f, 0.f, 0.f, 0.f, 0.f, 0.f, 0.f, 0.f};
  return z;
}

// ---------------- fp32 -> bf16 convert ----------------
__global__ void f2bf_kernel(const float* __restrict__ in, __bf16* __restrict__ out, size_t n) {
  size_t i = ((size_t)blockIdx.x * blockDim.x + threadIdx.x) * 4;
  #pragma unroll
  for (int k = 0; k < 4; k++) {
    size_t j = i + k;
    if (j < n) out[j] = (__bf16)in[j];
  }
}

// ---------------- transpose W[K,N] fp32 -> Wt[N,K] bf16 ----------------
__global__ __launch_bounds__(256) void transpose_w_kernel(const float* __restrict__ W,
                                                          __bf16* __restrict__ Wt,
                                                          int Kd, int N) {
  __shared__ float tile[32][33];
  int n0 = blockIdx.x * 32, k0 = blockIdx.y * 32;
  int tx = threadIdx.x, ty = threadIdx.y;
  #pragma unroll
  for (int i = ty; i < 32; i += 8)
    tile[i][tx] = W[(size_t)(k0 + i) * N + n0 + tx];
  __syncthreads();
  #pragma unroll
  for (int i = ty; i < 32; i += 8)
    Wt[(size_t)(n0 + i) * Kd + k0 + tx] = (__bf16)tile[tx][i];
}

// ---------------- WMMA GEMM: C[M,N] f32 = A[M,Kd] bf16 * Bt[N,Kd]^T bf16 ----------------
// block tile 128x128, BK=32, 256 threads = 8 waves (4 m x 2 n), wave tile 32x64.
// Double-buffered LDS staged with gfx1250 async global->LDS copies (ASYNCcnt).
__global__ __launch_bounds__(256) void gemm_bf16_kernel(const __bf16* __restrict__ A,
                                                        const __bf16* __restrict__ Bt,
                                                        float* __restrict__ C,
                                                        int M, int N, int Kd) {
  __shared__ __bf16 sA[2][128 * 32];
  __shared__ __bf16 sB[2][128 * 32];
  const int tid  = threadIdx.x;
  const int lane = tid & 31, wid = tid >> 5;
  const int wm   = wid >> 1, wn = wid & 1;
  const int half = lane >> 4, r = lane & 15;
  const int m0 = blockIdx.y * 128, n0 = blockIdx.x * 128;

  v8f acc[2][4];
  #pragma unroll
  for (int i = 0; i < 2; i++)
    #pragma unroll
    for (int j = 0; j < 4; j++) acc[i][j] = zero8();

  // stage one 128x32 A tile + 128x32 B tile into buffer `buf`
  auto stage = [&](int buf, int kt) {
    #pragma unroll
    for (int i = 0; i < 2; i++) {
      int idx = tid + i * 256;                 // 512 x 16B chunks per matrix
      int rw  = idx >> 2;
      int cv  = (idx & 3) << 3;                // element offset rw*32+cv == idx*8
      async_cp16(&sA[buf][idx * 8], &A[(size_t)(m0 + rw) * Kd + kt + cv]);
      async_cp16(&sB[buf][idx * 8], &Bt[(size_t)(n0 + rw) * Kd + kt + cv]);
    }
  };

  int cur = 0;
  stage(0, 0);

  for (int kt = 0; kt < Kd; kt += 32) {
    async_wait0();        // my async copies into buffer `cur` have landed
    __syncthreads();      // everyone's copies landed; prev reads of other buffer done
    if (kt + 32 < Kd) stage(cur ^ 1, kt + 32);

    const __bf16* pA = sA[cur];
    const __bf16* pB = sB[cur];
    v16bf af[2], bfr[4];
    #pragma unroll
    for (int i = 0; i < 2; i++)
      af[i] = load_frag16(&pA[(wm * 32 + i * 16 + r) * 32 + half * 8]);
    #pragma unroll
    for (int j = 0; j < 4; j++)
      bfr[j] = load_frag16(&pB[(wn * 64 + j * 16 + r) * 32 + half * 8]);

    #pragma unroll
    for (int i = 0; i < 2; i++)
      #pragma unroll
      for (int j = 0; j < 4; j++)
        acc[i][j] = __builtin_amdgcn_wmma_f32_16x16x32_bf16(
            false, af[i], false, bfr[j], (short)0, acc[i][j], false, false);
    cur ^= 1;
  }

  // C layout: lane row = vr + 8*half, lane col = lane&15
  #pragma unroll
  for (int i = 0; i < 2; i++) {
    #pragma unroll
    for (int j = 0; j < 4; j++) {
      int row = m0 + wm * 32 + i * 16 + 8 * half;
      int col = n0 + wn * 64 + j * 16 + r;
      const float* av = reinterpret_cast<const float*>(&acc[i][j]);
      #pragma unroll
      for (int vr = 0; vr < 8; vr++)
        C[(size_t)(row + vr) * N + col] = av[vr];
    }
  }
}

// ---------------- RMSNorm + RoPE -> bf16 [B, nh, T, Hd] ----------------
__global__ __launch_bounds__(128) void rmsnorm_rope_kernel(const float* __restrict__ in,
                                                           const float* __restrict__ w,
                                                           const float* __restrict__ sn,
                                                           const float* __restrict__ cs,
                                                           __bf16* __restrict__ out,
                                                           int nh) {
  __shared__ float red[4];
  __shared__ float xs[Hd];
  const int bt = blockIdx.x;            // b*T + t
  const int head = blockIdx.y;
  const int t = bt % Tq, b = bt / Tq;
  const int tid = threadIdx.x;          // 0..127

  float x = in[(size_t)bt * (nh * Hd) + head * Hd + tid];
  float ss = x * x;
  #pragma unroll
  for (int m = 16; m >= 1; m >>= 1) ss += __shfl_xor(ss, m, 32);
  if ((tid & 31) == 0) red[tid >> 5] = ss;
  __syncthreads();
  ss = red[0] + red[1] + red[2] + red[3];
  float xn = x * rsqrtf(ss * (1.0f / Hd) + RMS_EPS) * w[tid];
  xs[tid] = xn;
  __syncthreads();

  int d = tid & 63;
  float s = sn[t * 64 + d], c = cs[t * 64 + d];
  float x1 = xs[2 * d], x2 = xs[2 * d + 1];
  float o = (tid < 64) ? (x1 * c - x2 * s) : (x1 * s + x2 * c);
  out[(((size_t)b * nh + head) * Tq + t) * Hd + tid] = (__bf16)o;
}

// ---------------- V: [B*T, K*Hd] f32 -> Vt[B, K, Hd, T] bf16 ----------------
__global__ __launch_bounds__(256) void v_transpose_kernel(const float* __restrict__ V,
                                                          __bf16* __restrict__ Vt) {
  __shared__ float tile[32][33];
  int bk = blockIdx.z;
  int b = bk / Kq, kh = bk % Kq;
  int t0 = blockIdx.x * 32, d0 = blockIdx.y * 32;
  int tx = threadIdx.x, ty = threadIdx.y;
  #pragma unroll
  for (int i = ty; i < 32; i += 8)
    tile[i][tx] = V[((size_t)(b * Tq + t0 + i)) * (Kq * Hd) + kh * Hd + d0 + tx];
  __syncthreads();
  #pragma unroll
  for (int i = ty; i < 32; i += 8)
    Vt[(((size_t)(b * Kq) + kh) * Hd + d0 + i) * Tq + t0 + tx] = (__bf16)tile[tx][i];
}

// ---------------- Flash attention (causal, GQA) ----------------
// grid (T/64, B*H), 128 threads = 4 waves, each wave owns 16 q rows.
__global__ __launch_bounds__(128) void fattn_kernel(const __bf16* __restrict__ Qa,
                                                    const __bf16* __restrict__ Ka,
                                                    const __bf16* __restrict__ Vt,
                                                    __bf16* __restrict__ Oa) {
  __shared__ __bf16 sP[4][16 * 32];
  const int tid  = threadIdx.x;
  const int lane = tid & 31, wave = tid >> 5;
  const int half = lane >> 4, r = lane & 15;
  const int bh = blockIdx.y;
  const int b = bh / Hq, h = bh % Hq;
  const int kvh = h / (Hq / Kq);
  const int qrow0 = blockIdx.x * 64 + wave * 16;

  const __bf16* Qp = Qa + ((size_t)(b * Hq + h)) * Tq * Hd;
  const __bf16* Kp = Ka + ((size_t)(b * Kq + kvh)) * Tq * Hd;
  const __bf16* Vp = Vt + ((size_t)(b * Kq + kvh)) * Hd * Tq;
  __bf16* myP = sP[wave];

  // Q fragments (A layout), kept resident: 4 K-steps of 32 over Hd=128
  v16bf qf[4];
  #pragma unroll
  for (int ks = 0; ks < 4; ks++)
    qf[ks] = load_frag16(Qp + (size_t)(qrow0 + r) * Hd + ks * 32 + half * 8);

  float mrow[8], lrow[8];
  v8f o[8];
  #pragma unroll
  for (int i = 0; i < 8; i++) { mrow[i] = -3.0e38f; lrow[i] = 0.f; o[i] = zero8(); }
  const float sc = 0.08838834764831845f;   // 1/sqrt(128)

  for (int kt0 = 0; kt0 < qrow0 + 16; kt0 += 32) {
    // prefetch next K tile rows (global_prefetch_b8)
    if (kt0 + 32 < qrow0 + 16) {
      __builtin_prefetch(Kp + (size_t)(kt0 + 32 + r) * Hd, 0, 1);
      __builtin_prefetch(Kp + (size_t)(kt0 + 48 + r) * Hd, 0, 1);
    }

    // ---- S = Q K^T for 16 x 32 tile (two 16x16 accumulators) ----
    v8f s0 = zero8(), s1 = zero8();
    #pragma unroll
    for (int ks = 0; ks < 4; ks++) {
      int t0 = kt0 + r;       if (t0 >= Tq) t0 = 0;   // masked anyway
      int t1 = kt0 + 16 + r;  if (t1 >= Tq) t1 = 0;
      v16bf kf0 = load_frag16(Kp + (size_t)t0 * Hd + ks * 32 + half * 8);
      s0 = __builtin_amdgcn_wmma_f32_16x16x32_bf16(false, qf[ks], false, kf0,
                                                   (short)0, s0, false, false);
      v16bf kf1 = load_frag16(Kp + (size_t)t1 * Hd + ks * 32 + half * 8);
      s1 = __builtin_amdgcn_wmma_f32_16x16x32_bf16(false, qf[ks], false, kf1,
                                                   (short)0, s1, false, false);
    }

    // ---- online softmax (row stats via half-wave shuffles) ----
    float* sp0 = reinterpret_cast<float*>(&s0);
    float* sp1 = reinterpret_cast<float*>(&s1);
    #pragma unroll
    for (int vr = 0; vr < 8; vr++) {
      int rowg = qrow0 + vr + 8 * half;
      float a0 = (kt0 + r      <= rowg) ? sp0[vr] * sc : -3.0e38f;
      float a1 = (kt0 + 16 + r <= rowg) ? sp1[vr] * sc : -3.0e38f;
      float pm = fmaxf(a0, a1);
      #pragma unroll
      for (int m = 8; m >= 1; m >>= 1) pm = fmaxf(pm, __shfl_xor(pm, m, 32));
      float mnew  = fmaxf(mrow[vr], pm);
      float alpha = __expf(mrow[vr] - mnew);
      float p0 = __expf(a0 - mnew);
      float p1 = __expf(a1 - mnew);
      float ps = p0 + p1;
      #pragma unroll
      for (int m = 8; m >= 1; m >>= 1) ps += __shfl_xor(ps, m, 32);
      lrow[vr] = lrow[vr] * alpha + ps;
      mrow[vr] = mnew;
      myP[(vr + 8 * half) * 32 + r]      = (__bf16)p0;
      myP[(vr + 8 * half) * 32 + 16 + r] = (__bf16)p1;
      #pragma unroll
      for (int d = 0; d < 8; d++)
        reinterpret_cast<float*>(&o[d])[vr] *= alpha;
    }

    // ---- O += P V : P reloaded from LDS in A layout, contract K=32 ----
    v16bf pf = load_frag16(myP + r * 32 + half * 8);
    int g0 = kt0 + half * 8;      if (g0 >= Tq) g0 = 0;
    int g1 = kt0 + 16 + half * 8; if (g1 >= Tq) g1 = 0;
    #pragma unroll
    for (int dt = 0; dt < 8; dt++) {
      int d = dt * 16 + r;
      v16bf vf;
      U4* vp = reinterpret_cast<U4*>(&vf);
      vp[0] = *reinterpret_cast<const U4*>(Vp + (size_t)d * Tq + g0);
      vp[1] = *reinterpret_cast<const U4*>(Vp + (size_t)d * Tq + g1);
      o[dt] = __builtin_amdgcn_wmma_f32_16x16x32_bf16(false, pf, false, vf,
                                                      (short)0, o[dt], false, false);
    }
  }

  // ---- normalize and store bf16 into [B*T, H*Hd] for the output GEMM ----
  #pragma unroll
  for (int dt = 0; dt < 8; dt++) {
    const float* ov = reinterpret_cast<const float*>(&o[dt]);
    #pragma unroll
    for (int vr = 0; vr < 8; vr++) {
      int t = qrow0 + vr + 8 * half;
      int dcol = dt * 16 + r;
      Oa[(size_t)(b * Tq + t) * (Hq * Hd) + h * Hd + dcol] = (__bf16)(ov[vr] / lrow[vr]);
    }
  }
}

// ---------------- host launch ----------------
extern "C" void kernel_launch(void* const* d_in, const int* in_sizes, int n_in,
                              void* d_out, int out_size, void* d_ws, size_t ws_size,
                              hipStream_t stream) {
  (void)in_sizes; (void)n_in; (void)out_size; (void)ws_size;
  const float* x   = (const float*)d_in[0];
  const float* Wq  = (const float*)d_in[1];
  const float* Wk  = (const float*)d_in[2];
  const float* Wv  = (const float*)d_in[3];
  const float* Wo  = (const float*)d_in[4];
  const float* qnw = (const float*)d_in[5];
  const float* knw = (const float*)d_in[6];
  const float* sn  = (const float*)d_in[7];
  const float* cs  = (const float*)d_in[8];
  float* out = (float*)d_out;

  char* ws = (char*)d_ws;
  size_t off = 0;
  auto alloc = [&](size_t bytes) -> char* {
    char* p = ws + off;
    off += (bytes + 255) & ~(size_t)255;
    return p;
  };
  __bf16* xb  = (__bf16*)alloc((size_t)BT * Dq * 2);            // 16 MB
  __bf16* Wqt = (__bf16*)alloc((size_t)Dq * (Hq * Hd) * 2);     //  8 MB
  __bf16* Wkt = (__bf16*)alloc((size_t)Dq * (Kq * Hd) * 2);     //  4 MB
  __bf16* Wvt = (__bf16*)alloc((size_t)Dq * (Kq * Hd) * 2);     //  4 MB
  __bf16* Wot = (__bf16*)alloc((size_t)(Hq * Hd) * Dq * 2);     //  8 MB
  float*  Qf  = (float*)alloc((size_t)BT * (Hq * Hd) * 4);      // 32 MB
  float*  Kf  = (float*)alloc((size_t)BT * (Kq * Hd) * 4);      // 16 MB
  float*  Vf  = (float*)alloc((size_t)BT * (Kq * Hd) * 4);      // 16 MB
  __bf16* Qa  = (__bf16*)alloc((size_t)BT * (Hq * Hd) * 2);     // 16 MB
  __bf16* Ka  = (__bf16*)alloc((size_t)BT * (Kq * Hd) * 2);     //  8 MB
  __bf16* Vtb = (__bf16*)alloc((size_t)BT * (Kq * Hd) * 2);     //  8 MB
  __bf16* Ao  = (__bf16*)Qf;  // alias: Qf is dead after rmsnorm_rope(q)

  // 1) x -> bf16
  {
    size_t n = (size_t)BT * Dq;
    size_t blocks = (n + 1023) / 1024;
    f2bf_kernel<<<dim3((unsigned)blocks), 256, 0, stream>>>(x, xb, n);
  }
  // 2) weight transposes (fp32 -> bf16, N x K)
  transpose_w_kernel<<<dim3((Hq * Hd) / 32, Dq / 32), dim3(32, 8), 0, stream>>>(Wq, Wqt, Dq, Hq * Hd);
  transpose_w_kernel<<<dim3((Kq * Hd) / 32, Dq / 32), dim3(32, 8), 0, stream>>>(Wk, Wkt, Dq, Kq * Hd);
  transpose_w_kernel<<<dim3((Kq * Hd) / 32, Dq / 32), dim3(32, 8), 0, stream>>>(Wv, Wvt, Dq, Kq * Hd);
  transpose_w_kernel<<<dim3(Dq / 32, (Hq * Hd) / 32), dim3(32, 8), 0, stream>>>(Wo, Wot, Hq * Hd, Dq);
  // 3) QKV projections
  gemm_bf16_kernel<<<dim3((Hq * Hd) / 128, BT / 128), 256, 0, stream>>>(xb, Wqt, Qf, BT, Hq * Hd, Dq);
  gemm_bf16_kernel<<<dim3((Kq * Hd) / 128, BT / 128), 256, 0, stream>>>(xb, Wkt, Kf, BT, Kq * Hd, Dq);
  gemm_bf16_kernel<<<dim3((Kq * Hd) / 128, BT / 128), 256, 0, stream>>>(xb, Wvt, Vf, BT, Kq * Hd, Dq);
  // 4) RMSNorm + RoPE (q, k)
  rmsnorm_rope_kernel<<<dim3(BT, Hq), 128, 0, stream>>>(Qf, qnw, sn, cs, Qa, Hq);
  rmsnorm_rope_kernel<<<dim3(BT, Kq), 128, 0, stream>>>(Kf, knw, sn, cs, Ka, Kq);
  // 5) V transpose to [B,K,Hd,T]
  v_transpose_kernel<<<dim3(Tq / 32, Hd / 32, Bq * Kq), dim3(32, 8), 0, stream>>>(Vf, Vtb);
  // 6) flash attention
  fattn_kernel<<<dim3(Tq / 64, Bq * Hq), 128, 0, stream>>>(Qa, Ka, Vtb, Ao);
  // 7) output projection -> fp32 out
  gemm_bf16_kernel<<<dim3(Dq / 128, BT / 128), 256, 0, stream>>>(Ao, Wot, out, BT, Dq, Dq);
}
